// ScaledDotProductAttention_85684597555799
// MI455X (gfx1250) — matrix operand order repaired and checked by reading, not compiled, not run on hardware
//
#include <hip/hip_runtime.h>
#include <hip/hip_bf16.h>

// ---------------------------------------------------------------------------
// Causal single-head attention, MI455X (gfx1250, wave32, WMMA bf16).
//   Phase 1: q = x@wq, k = x@wk, v = x@wv   (stored bf16; v transposed)
//   Phase 2: flash attention (online softmax), fp32 accumulate, fp32 out
// ---------------------------------------------------------------------------

typedef __bf16        v16bf __attribute__((ext_vector_type(16)));
typedef float         v8f   __attribute__((ext_vector_type(8)));
typedef unsigned int  v4u   __attribute__((ext_vector_type(4)));

union BV { v16bf v; v4u q[2]; };

#define NBATCH 8
#define NT     2048
#define NDM    1024
#define NH     64
#define NBT    (NBATCH * NT)   // 16384 rows total

// ds_swizzle group-of-32 XOR butterfly (masks < 16 stay within each 16-lane half)
#define SWZF(val, m) __int_as_float(__builtin_amdgcn_ds_swizzle( \
    __float_as_int((val)), (0x1f | ((m) << 10))))

__device__ inline v8f v8f_zero() {
  v8f z;
#pragma unroll
  for (int i = 0; i < 8; ++i) z[i] = 0.0f;
  return z;
}

// ---------------------------------------------------------------------------
// Phase 1: QKV projection.  256 threads = 8 waves; wave w owns 16 rows.
// Weight K-chunks (32x64) staged to LDS transposed as bf16 so B-operands are
// contiguous ds_load_b128 in WMMA B layout.
// ---------------------------------------------------------------------------
__global__ __launch_bounds__(256) void qkv_proj_k(
    const float* __restrict__ x,
    const float* __restrict__ wq, const float* __restrict__ wk,
    const float* __restrict__ wv,
    __bf16* __restrict__ qB, __bf16* __restrict__ kB, __bf16* __restrict__ vT)
{
  __shared__ __align__(16) __bf16 smem[3][64][40];  // [mat][N][K(+pad)]

  const int tid  = threadIdx.x;
  const int lane = tid & 31;
  const int wid  = tid >> 5;
  const int m16  = lane & 15;
  const int hi   = lane >> 4;           // 0: lanes 0-15, 1: lanes 16-31
  const int row0 = blockIdx.x * 128 + wid * 16;

  v8f acc[3][4];
#pragma unroll
  for (int m = 0; m < 3; ++m)
#pragma unroll
    for (int n = 0; n < 4; ++n) acc[m][n] = v8f_zero();

  const float* wsrc3[3] = { wq, wk, wv };
  const float* xrow = x + (size_t)(row0 + m16) * NDM + hi * 8;

  for (int kk = 0; kk < 32; ++kk) {
    // --- stage transposed bf16 weight chunk: smem[m][n][k] = w[kk*32+k][n]
#pragma unroll
    for (int m = 0; m < 3; ++m) {
      const float* wsrc = wsrc3[m] + (size_t)kk * 2048;
#pragma unroll
      for (int i = 0; i < 8; ++i) {
        int f = i * 256 + tid;                 // 0..2047 coalesced read
        smem[m][f & 63][f >> 6] = (__bf16)wsrc[f];
      }
    }
    __syncthreads();

    // --- A operand: x tile 16x32 in WMMA A layout (row = lane%16)
    BV a;
#pragma unroll
    for (int j = 0; j < 8; ++j) {
      a.v[j]     = (__bf16)xrow[j];            // K = klo + 0..7
      a.v[8 + j] = (__bf16)xrow[16 + j];       // K = klo + 16..23
    }

    // --- 12 WMMAs: 3 matrices x 4 N-tiles
#pragma unroll
    for (int m = 0; m < 3; ++m) {
#pragma unroll
      for (int n = 0; n < 4; ++n) {
        BV bb;
        const __bf16* bp = &smem[m][n * 16 + m16][hi * 8];
        bb.q[0] = *(const v4u*)bp;             // K = kbase + 0..7
        bb.q[1] = *(const v4u*)(bp + 16);      // K = kbase + 16..23
        acc[m][n] = __builtin_amdgcn_wmma_f32_16x16x32_bf16(
            false, a.v, false, bb.v, (short)0, acc[m][n], false, false);
      }
    }
    __syncthreads();
    xrow += 32;
  }

  // --- epilogue: q,k row-major bf16; v transposed vT[b][h][t]
#pragma unroll
  for (int r = 0; r < 8; ++r) {
    const int t    = row0 + r + hi * 8;
    const int bidx = t >> 11;
    const int tt   = t & 2047;
#pragma unroll
    for (int n = 0; n < 4; ++n) {
      const int h = n * 16 + m16;
      qB[(size_t)t * NH + h] = (__bf16)acc[0][n][r];
      kB[(size_t)t * NH + h] = (__bf16)acc[1][n][r];
      vT[((size_t)bidx * NH + h) * NT + tt] = (__bf16)acc[2][n][r];
    }
  }
}

// ---------------------------------------------------------------------------
// Phase 2: flash attention. One wave per 16-query tile; 32-key blocks.
// ---------------------------------------------------------------------------
__global__ __launch_bounds__(256) void attn_k(
    const __bf16* __restrict__ qB, const __bf16* __restrict__ kB,
    const __bf16* __restrict__ vT, float* __restrict__ out)
{
  __shared__ __align__(16) __bf16 ldsp[8][16][32];  // per-wave P tile (C->A relayout)

  const int tid  = threadIdx.x;
  const int lane = tid & 31;
  const int wid  = tid >> 5;
  const int m16  = lane & 15;
  const int hi   = lane >> 4;

  const int ti = blockIdx.x * 8 + wid;     // 0..1023 query tiles
  const int b  = ti & 7;
  const int q0 = (ti >> 3) * 16;

  // Q in WMMA A layout for both head-dim chunks (K = 0..31 and 32..63)
  BV aq[2];
  {
    const __bf16* qrow = qB + (size_t)(b * NT + q0 + m16) * NH + hi * 8;
#pragma unroll
    for (int s = 0; s < 2; ++s) {
      aq[s].q[0] = *(const v4u*)(qrow + s * 32);
      aq[s].q[1] = *(const v4u*)(qrow + s * 32 + 16);
    }
  }

  v8f acc[4];
#pragma unroll
  for (int n = 0; n < 4; ++n) acc[n] = v8f_zero();
  float mrow[8], lrow[8];
#pragma unroll
  for (int r = 0; r < 8; ++r) { mrow[r] = -__builtin_inff(); lrow[r] = 0.0f; }

  const float k1 = 0.125f * 1.44269504088896340736f;  // 1/sqrt(64) * log2(e)
  const int nb = (q0 >> 5) + 1;                       // causal key-block count

  for (int kb = 0; kb < nb; ++kb) {
    const int key0 = kb * 32;

    // --- S = Q @ K^T : two 16x16 C tiles, 2 WMMA steps each over head dim
    v8f S[2];
#pragma unroll
    for (int t2 = 0; t2 < 2; ++t2) {
      v8f c = v8f_zero();
      const __bf16* krow =
          kB + (size_t)(b * NT + key0 + t2 * 16 + m16) * NH + hi * 8;
#pragma unroll
      for (int s = 0; s < 2; ++s) {
        BV bk;
        bk.q[0] = *(const v4u*)(krow + s * 32);
        bk.q[1] = *(const v4u*)(krow + s * 32 + 16);
        c = __builtin_amdgcn_wmma_f32_16x16x32_bf16(
            false, aq[s].v, false, bk.v, (short)0, c, false, false);
      }
      S[t2] = c;
    }

    // --- online softmax (rows striped across 16-lane halves of C layout)
    const bool lastb = (kb == nb - 1);
#pragma unroll
    for (int r = 0; r < 8; ++r) {
      float s0 = S[0][r] * k1;
      float s1 = S[1][r] * k1;
      const int qi = q0 + r + hi * 8;
      if (lastb) {
        if (key0 + m16 > qi)      s0 = -__builtin_inff();
        if (key0 + 16 + m16 > qi) s1 = -__builtin_inff();
      }
      float vmax = fmaxf(s0, s1);
      vmax = fmaxf(vmax, SWZF(vmax, 1));
      vmax = fmaxf(vmax, SWZF(vmax, 2));
      vmax = fmaxf(vmax, SWZF(vmax, 4));
      vmax = fmaxf(vmax, SWZF(vmax, 8));
      const float mnew  = fmaxf(mrow[r], vmax);
      const float p0    = __builtin_amdgcn_exp2f(s0 - mnew);
      const float p1    = __builtin_amdgcn_exp2f(s1 - mnew);
      const float alpha = __builtin_amdgcn_exp2f(mrow[r] - mnew);
      mrow[r] = mnew;
      float ps = p0 + p1;
      ps += SWZF(ps, 1);
      ps += SWZF(ps, 2);
      ps += SWZF(ps, 4);
      ps += SWZF(ps, 8);
      lrow[r] = lrow[r] * alpha + ps;
#pragma unroll
      for (int n = 0; n < 4; ++n) acc[n][r] = acc[n][r] * alpha;
      const int mm = r + hi * 8;
      ldsp[wid][mm][m16]      = (__bf16)p0;   // C layout -> LDS row-major
      ldsp[wid][mm][16 + m16] = (__bf16)p1;
    }
    __asm__ volatile("s_wait_dscnt 0" ::: "memory");

    // --- P back out of LDS in WMMA A layout (16x32)
    BV ap;
    {
      const __bf16* p = &ldsp[wid][m16][hi * 8];
      ap.q[0] = *(const v4u*)p;
      ap.q[1] = *(const v4u*)(p + 16);
    }

    // --- acc += P @ V (vT rows contiguous along keys -> direct B layout)
#pragma unroll
    for (int n = 0; n < 4; ++n) {
      BV bv;
      const __bf16* vp =
          vT + ((size_t)b * NH + n * 16 + m16) * NT + key0 + hi * 8;
      bv.q[0] = *(const v4u*)vp;
      bv.q[1] = *(const v4u*)(vp + 16);
      acc[n] = __builtin_amdgcn_wmma_f32_16x16x32_bf16(
          false, ap.v, false, bv.v, (short)0, acc[n], false, false);
    }
  }

  // --- normalize and store fp32 output [B,T,H]
#pragma unroll
  for (int r = 0; r < 8; ++r) {
    const float inv = 1.0f / lrow[r];
    const int t = q0 + r + hi * 8;
    float* orow = out + (size_t)(b * NT + t) * NH;
#pragma unroll
    for (int n = 0; n < 4; ++n) orow[n * 16 + m16] = acc[n][r] * inv;
  }
}

// ---------------------------------------------------------------------------
extern "C" void kernel_launch(void* const* d_in, const int* in_sizes, int n_in,
                              void* d_out, int out_size, void* d_ws, size_t ws_size,
                              hipStream_t stream) {
  (void)in_sizes; (void)n_in; (void)out_size; (void)ws_size;
  const float* x  = (const float*)d_in[0];
  const float* wq = (const float*)d_in[1];
  const float* wk = (const float*)d_in[2];
  const float* wv = (const float*)d_in[3];

  __bf16* qB = (__bf16*)d_ws;                       // 2 MB
  __bf16* kB = qB + (size_t)NBT * NH;               // 2 MB
  __bf16* vT = kB + (size_t)NBT * NH;               // 2 MB (transposed [b][h][t])

  qkv_proj_k<<<dim3(NBT / 128), dim3(256), 0, stream>>>(x, wq, wk, wv, qB, kB, vT);
  attn_k<<<dim3((NBATCH * (NT / 16)) / 8), dim3(256), 0, stream>>>(qB, kB, vT,
                                                                   (float*)d_out);
}
